// MoHSparseAttentionSimple_20985210208479
// MI455X (gfx1250) — matrix operand order repaired and checked by reading, not compiled, not run on hardware
//
#include <hip/hip_runtime.h>
#include <hip/hip_bf16.h>

// MoH sparse attention for gfx1250 (MI455X): bf16 WMMA + TDM tile staging.
// wave32, 256-thread blocks = 8 waves, WMMA 16x16x32 bf16, f32 accumulate.

typedef __attribute__((ext_vector_type(8)))  float        v8f;
typedef __attribute__((ext_vector_type(16))) __bf16       bf16x16;
typedef __attribute__((ext_vector_type(4)))  unsigned int u32x4;
typedef __attribute__((ext_vector_type(4)))  int          i32x4;
typedef __attribute__((ext_vector_type(8)))  int          i32x8;

static constexpr int NH   = 32;    // total heads
static constexpr int TOPK = 8;     // active heads
static constexpr int HD   = 128;   // head dim
static constexpr int EMB  = 1024;  // embed dim
static constexpr int NB   = 2;     // batch
static constexpr int SEQ  = 2048;  // sequence
#define ATT_SCALE 0.08838834764831845f
#define ENT_W_ 0.01f

// ---------- Tensor Data Mover: 2D bf16 tile -> LDS (with row padding) ------
// Descriptor per CDNA5 ISA. Groups 2/3 zero (2D).
__device__ __forceinline__ void tdm_load_2d_bf16(
    unsigned lds_addr, const __bf16* gptr,
    unsigned tile_w, unsigned tile_h,
    unsigned row_stride_elems, unsigned tensor_rows,
    unsigned pad_interval, unsigned pad_amount) {
  unsigned long long ga = (unsigned long long)(size_t)gptr;
  u32x4 g0;
  g0[0] = 1u;                                        // count=1 (valid user D#)
  g0[1] = lds_addr;                                  // LDS byte address
  g0[2] = (unsigned)(ga & 0xffffffffu);              // global addr lo
  g0[3] = (unsigned)((ga >> 32) & 0x01ffffffu) | (2u << 30);  // addr hi | type=2
  i32x8 g1;
  // data_size=1 (2B) | pad_enable | pad_interval | pad_amount
  g1[0] = (int)((1u << 16) | (1u << 20) | (pad_interval << 22) | (pad_amount << 25));
  g1[1] = (int)((row_stride_elems & 0xffffu) << 16);         // tensor_dim0[15:0]
  g1[2] = (int)((row_stride_elems >> 16) & 0xffffu) |
          (int)((tensor_rows & 0xffffu) << 16);              // td0 hi | td1 lo
  g1[3] = (int)((tensor_rows >> 16) & 0xffffu) |
          (int)(tile_w << 16);                               // td1 hi | tile_dim0
  g1[4] = (int)(tile_h & 0xffffu);                           // tile_dim1 (dim2=0)
  g1[5] = (int)row_stride_elems;                             // dim0_stride lo32
  g1[6] = 0;                                                 // stride hi | d1stride lo
  g1[7] = 0;
  i32x4 gz = {0, 0, 0, 0};
#if __clang_major__ >= 23
  i32x8 gz8 = {0, 0, 0, 0, 0, 0, 0, 0};
  __builtin_amdgcn_tensor_load_to_lds(g0, g1, gz, gz, gz8, 0);
#else
  __builtin_amdgcn_tensor_load_to_lds(g0, g1, gz, gz, 0);
#endif
}

// ---------- lane reductions (wave32) ----------
__device__ __forceinline__ float h16max(float v) {
#pragma unroll
  for (int m = 1; m < 16; m <<= 1) v = fmaxf(v, __shfl_xor(v, m, 32));
  return v;
}
__device__ __forceinline__ float h16sum(float v) {
#pragma unroll
  for (int m = 1; m < 16; m <<= 1) v += __shfl_xor(v, m, 32);
  return v;
}
__device__ __forceinline__ float w32max(float v) {
#pragma unroll
  for (int m = 1; m < 32; m <<= 1) v = fmaxf(v, __shfl_xor(v, m, 32));
  return v;
}
__device__ __forceinline__ float w32sum(float v) {
#pragma unroll
  for (int m = 1; m < 32; m <<= 1) v += __shfl_xor(v, m, 32);
  return v;
}

// ---------- utility kernels ----------
__global__ void zero_f32(float* p, int n) {
  int i = blockIdx.x * blockDim.x + threadIdx.x;
  if (i < n) p[i] = 0.0f;
}

__global__ void f32_to_bf16(const float* __restrict__ in,
                            __bf16* __restrict__ out, int n) {
  int i = blockIdx.x * blockDim.x + threadIdx.x;
  if (i < n) out[i] = (__bf16)in[i];
}

// transpose [K][N] f32 -> [N][K] bf16 (tiled through LDS)
__global__ __launch_bounds__(256) void transpose_f32_to_bf16(
    const float* __restrict__ in, __bf16* __restrict__ out, int K, int N) {
  __shared__ float tile[32][33];
  const int n0 = blockIdx.x * 32;
  const int k0 = blockIdx.y * 32;
  const int tx = threadIdx.x & 31;
  const int ty = threadIdx.x >> 5;  // 0..7
  for (int i = ty; i < 32; i += 8)
    tile[i][tx] = in[(size_t)(k0 + i) * N + n0 + tx];
  __syncthreads();
  for (int i = ty; i < 32; i += 8)
    out[(size_t)(n0 + i) * K + k0 + tx] = (__bf16)tile[tx][i];
}

// ---------- router: one wave per token ----------
__global__ __launch_bounds__(32) void router_kernel(
    const float* __restrict__ x, const float* __restrict__ Wr,
    int* __restrict__ idx_out, float* __restrict__ wts_out,
    float* __restrict__ p_acc, float* __restrict__ f_acc,
    float* __restrict__ ent_acc) {
  __shared__ float xs[EMB];
  const int tok  = blockIdx.x;   // b*SEQ + s
  const int b    = tok / SEQ;
  const int lane = threadIdx.x;  // 0..31 == head id
  for (int e = lane; e < EMB; e += 32) xs[e] = x[(size_t)tok * EMB + e];
  __syncthreads();
  float logit = 0.0f;
  for (int e = 0; e < EMB; ++e) logit = fmaf(xs[e], Wr[e * NH + lane], logit);

  float mx = w32max(logit);
  float ex = __expf(logit - mx);
  float sm = w32sum(ex);
  float p  = ex / sm;
  float ent = w32sum(-p * __logf(p + 1e-8f));

  float v = logit;
  float topv[TOPK]; int topi[TOPK];
#pragma unroll
  for (int k = 0; k < TOPK; ++k) {
    float mv = w32max(v);
    unsigned long long bal = __ballot(v == mv);
    int winner = __ffsll(bal) - 1;
    topv[k] = mv; topi[k] = winner;
    if (lane == winner) v = -3.0e38f;
  }
  float wsum = 0.0f;
#pragma unroll
  for (int k = 0; k < TOPK; ++k) wsum += __expf(topv[k] - topv[0]);
  if (lane < TOPK) {
    idx_out[tok * TOPK + lane] = topi[lane];
    wts_out[tok * TOPK + lane] = __expf(topv[lane] - topv[0]) / wsum;
  }
  atomicAdd(&p_acc[b * NH + lane], p);
  if (lane == 0) {
    atomicAdd(ent_acc, ent);
    atomicAdd(&f_acc[b * NH + topi[0]], 1.0f);
  }
}

__global__ void aux_finalize(const float* __restrict__ p_acc,
                             const float* __restrict__ f_acc,
                             const float* __restrict__ ent_acc,
                             float* __restrict__ out_tail) {
  int t = threadIdx.x;
  if (t < NB * NH) out_tail[t] = 0.0f;  // head_counts (int32 zeros == f32 zeros)
  if (t == 0) {
    float bal = 0.0f;
    for (int i = 0; i < NB * NH; ++i)
      bal += (f_acc[i] / (float)SEQ) * (p_acc[i] / (float)SEQ);
    bal = bal * (float)NH / (float)NB;
    float ent = ent_acc[0] / (float)(NB * SEQ);
    out_tail[NB * NH] = bal - ENT_W_ * ent;
  }
}

// ---------- WMMA GEMM: C[MxN] = A[MxK] * Bt[NxK]^T, bf16 in, CT out --------
// Both operands K-contiguous; tiles staged to LDS by the Tensor Data Mover.
// Block tile 128x128, 8 waves (4 along M x 2 along N), wave tile 32x64.
template <typename CT>
__global__ __launch_bounds__(256) void gemm_bf16(
    const __bf16* __restrict__ A, const __bf16* __restrict__ Bt,
    CT* __restrict__ C, int M, int N, int Kd) {
  __shared__ __bf16 As[128][40];  // [m][k]  (40-elem row via TDM pad)
  __shared__ __bf16 Bs[128][40];  // [n][k]
  const int bm = blockIdx.y * 128;
  const int bn = blockIdx.x * 128;
  const int tid  = threadIdx.x;
  const int lane = tid & 31;
  const int wv   = tid >> 5;
  const int wm   = (wv & 3) * 32;
  const int wn   = (wv >> 2) * 64;

  v8f acc[2][4];
#pragma unroll
  for (int i = 0; i < 2; ++i)
#pragma unroll
    for (int j = 0; j < 4; ++j)
#pragma unroll
      for (int e = 0; e < 8; ++e) acc[i][j][e] = 0.0f;

  const int arow = lane & 15;
  const int ka0  = (lane < 16) ? 0 : 16;  // A-frag K bases (contiguous k = 16h + s)
  const int ka1  = ka0 + 8;
  const int kb   = (lane < 16) ? 0 : 16;  // B-frag K base

  const unsigned as_lds = (unsigned)(size_t)&As[0][0];
  const unsigned bs_lds = (unsigned)(size_t)&Bs[0][0];

  for (int k0 = 0; k0 < Kd; k0 += 32) {
    __syncthreads();  // previous iteration's consumers done
    if (wv == 0) {
      // 128x32 bf16 tiles; rows padded 32->40 elems (pad 16 dw interval, 4 dw)
      tdm_load_2d_bf16(as_lds, A  + (size_t)bm * Kd + k0, 32, 128,
                       (unsigned)Kd, (unsigned)M, 3, 3);
      tdm_load_2d_bf16(bs_lds, Bt + (size_t)bn * Kd + k0, 32, 128,
                       (unsigned)Kd, (unsigned)N, 3, 3);
      __builtin_amdgcn_s_wait_tensorcnt((short)0);
    }
    __syncthreads();

    bf16x16 af[2], bf[4];
#pragma unroll
    for (int i = 0; i < 2; ++i) {
      const __bf16* p0 = &As[wm + i * 16 + arow][ka0];
      const __bf16* p1 = &As[wm + i * 16 + arow][ka1];
#pragma unroll
      for (int j = 0; j < 8; ++j) { af[i][j] = p0[j]; af[i][8 + j] = p1[j]; }
    }
#pragma unroll
    for (int i = 0; i < 4; ++i) {
      const __bf16* p0 = &Bs[wn + i * 16 + arow][kb];
#pragma unroll
      for (int j = 0; j < 16; ++j) bf[i][j] = p0[j];
    }
#pragma unroll
    for (int i = 0; i < 2; ++i)
#pragma unroll
      for (int j = 0; j < 4; ++j)
        acc[i][j] = __builtin_amdgcn_wmma_f32_16x16x32_bf16(
            false, af[i], false, bf[j], (short)0, acc[i][j], false, false);
  }

  const int cn = lane & 15;
  const int rb = (lane < 16) ? 0 : 8;  // C/D layout: vgpr j -> row j / j+8
#pragma unroll
  for (int i = 0; i < 2; ++i)
#pragma unroll
    for (int j = 0; j < 4; ++j)
#pragma unroll
      for (int e = 0; e < 8; ++e) {
        int row = bm + wm + i * 16 + rb + e;
        int col = bn + wn + j * 16 + cn;
        C[(size_t)row * N + col] = (CT)acc[i][j][e];
      }
}

// ---------- gather selected heads (+RoPE for q/k) ----------
template <bool ROPE>
__global__ __launch_bounds__(256) void gather_heads(
    const __bf16* __restrict__ all,  // [B*S][NH*HD]
    const int* __restrict__ idx,     // [B*S][TOPK]
    __bf16* __restrict__ out) {      // [B][TOPK][S][HD]
  int gid = blockIdx.x * blockDim.x + threadIdx.x;
  int d   = gid & 127;
  int k   = (gid >> 7) & 7;
  int tok = gid >> 10;
  if (tok >= NB * SEQ) return;
  int b = tok / SEQ;
  int s = tok - b * SEQ;
  int h = idx[tok * TOPK + k];
  const __bf16* src = all + (size_t)tok * (NH * HD) + h * HD;
  float val = (float)src[d];
  if (ROPE) {
    int j = d & 63;
    float inv = __expf(-(float)j * (9.210340371976184f / 64.0f));
    float ang = (float)s * inv;
    float sn, cs;
    __sincosf(ang, &sn, &cs);
    float other = (d < 64) ? -(float)src[d + 64] : (float)src[d - 64];
    val = val * cs + other * sn;
  }
  out[(((size_t)b * TOPK + k) * SEQ + s) * HD + d] = (__bf16)val;
}

// V gathered transposed: [B][TOPK][HD][S] so the flash V tile is a plain 2D tile
__global__ __launch_bounds__(256) void gather_v(
    const __bf16* __restrict__ all, const int* __restrict__ idx,
    __bf16* __restrict__ out) {
  int gid = blockIdx.x * blockDim.x + threadIdx.x;
  int s = gid & (SEQ - 1);
  int d = (gid >> 11) & (HD - 1);
  int k = (gid >> 18) & (TOPK - 1);
  int b = gid >> 21;
  if (b >= NB) return;
  int tok = b * SEQ + s;
  int h = idx[tok * TOPK + k];
  out[(((size_t)b * TOPK + k) * HD + d) * SEQ + s] =
      all[(size_t)tok * (NH * HD) + h * HD + d];
}

// ---------- flash attention: block = (128 q rows, one (b,slot)) ----------
__global__ __launch_bounds__(256) void flash_attn(
    const __bf16* __restrict__ qm,  // [B][TOPK][S][HD]
    const __bf16* __restrict__ km,  // [B][TOPK][S][HD]
    const __bf16* __restrict__ vm,  // [B][TOPK][HD][S]
    const float* __restrict__ wts,  // [B*S][TOPK]
    __bf16* __restrict__ aout) {    // [B][S][TOPK*HD]
  __shared__ __bf16 Qs[128][136];   // [qrow][d]  (row 128 -> 136 via TDM pad)
  __shared__ __bf16 Ks[32][136];    // [t][d]
  __shared__ __bf16 Vs[128][40];    // [d][t]
  __shared__ __bf16 Ps[8][16][40];  // per-wave P staging

  const int qb   = blockIdx.x * 128;
  const int ks   = blockIdx.y;  // slot
  const int b    = blockIdx.z;
  const int lane = threadIdx.x & 31;
  const int wv   = threadIdx.x >> 5;

  const __bf16* qbase = qm + ((size_t)b * TOPK + ks) * SEQ * HD;
  const __bf16* kbase = km + ((size_t)b * TOPK + ks) * SEQ * HD;
  const __bf16* vbase = vm + ((size_t)b * TOPK + ks) * HD * SEQ;

  if (wv == 0) {
    // 128x128 Q tile; rows padded 128->136 (interval 64 dw, pad 4 dw)
    tdm_load_2d_bf16((unsigned)(size_t)&Qs[0][0], qbase + (size_t)qb * HD,
                     128, 128, HD, SEQ, 5, 3);
  }

  v8f Oacc[8];
#pragma unroll
  for (int dt = 0; dt < 8; ++dt)
#pragma unroll
    for (int e = 0; e < 8; ++e) Oacc[dt][e] = 0.0f;
  float m_i[8], l_i[8];
#pragma unroll
  for (int j = 0; j < 8; ++j) { m_i[j] = -1e30f; l_i[j] = 0.0f; }

  const int qr   = qb + wv * 16;  // this wave's q-row base
  const int arow = lane & 15;
  const int ka0  = (lane < 16) ? 0 : 16;
  const int ka1  = ka0 + 8;
  const int kb   = (lane < 16) ? 0 : 16;
  const int rb   = (lane < 16) ? 0 : 8;

  for (int t0 = 0; t0 < qb + 128; t0 += 32) {
    __syncthreads();
    if (wv == 0) {
      tdm_load_2d_bf16((unsigned)(size_t)&Ks[0][0], kbase + (size_t)t0 * HD,
                       128, 32, HD, SEQ, 5, 3);
      tdm_load_2d_bf16((unsigned)(size_t)&Vs[0][0], vbase + t0,
                       32, 128, SEQ, HD, 3, 3);
      __builtin_amdgcn_s_wait_tensorcnt((short)0);  // also covers Q tile
    }
    __syncthreads();
    if (t0 > qr + 15) continue;  // wave-uniform: fully masked chunk

    // S = Q * K^T  (16x32 per wave, 2 n-tiles, 4 k-steps over D=128)
    v8f s0, s1;
#pragma unroll
    for (int e = 0; e < 8; ++e) { s0[e] = 0.0f; s1[e] = 0.0f; }
#pragma unroll
    for (int d0 = 0; d0 < 128; d0 += 32) {
      bf16x16 aq;
      const __bf16* p0 = &Qs[wv * 16 + arow][d0 + ka0];
      const __bf16* p1 = &Qs[wv * 16 + arow][d0 + ka1];
#pragma unroll
      for (int j = 0; j < 8; ++j) { aq[j] = p0[j]; aq[8 + j] = p1[j]; }
      bf16x16 b0, b1;
      const __bf16* q0 = &Ks[arow][d0 + kb];
      const __bf16* q1 = &Ks[16 + arow][d0 + kb];
#pragma unroll
      for (int j = 0; j < 16; ++j) { b0[j] = q0[j]; b1[j] = q1[j]; }
      s0 = __builtin_amdgcn_wmma_f32_16x16x32_bf16(false, aq, false, b0,
                                                   (short)0, s0, false, false);
      s1 = __builtin_amdgcn_wmma_f32_16x16x32_bf16(false, aq, false, b1,
                                                   (short)0, s1, false, false);
    }

    // scale + causal mask + online softmax (rows live in lane-halves)
    float alpha[8];
    const int col0 = t0 + arow;
#pragma unroll
    for (int j = 0; j < 8; ++j) {
      int row = qr + rb + j;
      float v0 = s0[j] * ATT_SCALE; if (col0 > row)      v0 = -1e30f;
      float v1 = s1[j] * ATT_SCALE; if (col0 + 16 > row) v1 = -1e30f;
      float rm = h16max(fmaxf(v0, v1));
      float mn = fmaxf(m_i[j], rm);
      float al = __expf(m_i[j] - mn);
      float p0 = __expf(v0 - mn), p1 = __expf(v1 - mn);
      l_i[j] = l_i[j] * al + h16sum(p0 + p1);
      m_i[j] = mn; alpha[j] = al;
      Ps[wv][rb + j][arow]      = (__bf16)p0;
      Ps[wv][rb + j][16 + arow] = (__bf16)p1;
    }
#pragma unroll
    for (int dt = 0; dt < 8; ++dt)
#pragma unroll
      for (int j = 0; j < 8; ++j) Oacc[dt][j] *= alpha[j];

    // O += P * V
    bf16x16 pf;
    {
      const __bf16* p0 = &Ps[wv][arow][ka0];
      const __bf16* p1 = &Ps[wv][arow][ka1];
#pragma unroll
      for (int j = 0; j < 8; ++j) { pf[j] = p0[j]; pf[8 + j] = p1[j]; }
    }
#pragma unroll
    for (int dt = 0; dt < 8; ++dt) {
      bf16x16 bv;
      const __bf16* pv = &Vs[dt * 16 + arow][kb];
#pragma unroll
      for (int j = 0; j < 16; ++j) bv[j] = pv[j];
      Oacc[dt] = __builtin_amdgcn_wmma_f32_16x16x32_bf16(
          false, pf, false, bv, (short)0, Oacc[dt], false, false);
    }
  }

  // epilogue: normalize, weight by router prob, store [B][S][TOPK*HD]
#pragma unroll
  for (int j = 0; j < 8; ++j) {
    int r = qr + rb + j;
    float hw = wts[((size_t)b * SEQ + r) * TOPK + ks];
    float sc = hw / l_i[j];
#pragma unroll
    for (int dt = 0; dt < 8; ++dt) {
      aout[((size_t)b * SEQ + r) * (TOPK * HD) + ks * HD + dt * 16 + arow] =
          (__bf16)(Oacc[dt][j] * sc);
    }
  }
}

// ---------- host orchestration ----------
extern "C" void kernel_launch(void* const* d_in, const int* in_sizes, int n_in,
                              void* d_out, int out_size, void* d_ws,
                              size_t ws_size, hipStream_t stream) {
  (void)in_sizes; (void)n_in; (void)out_size; (void)ws_size;
  const float* x  = (const float*)d_in[0];
  const float* Wq = (const float*)d_in[1];
  const float* Wk = (const float*)d_in[2];
  const float* Wv = (const float*)d_in[3];
  const float* Wr = (const float*)d_in[4];
  const float* Wo = (const float*)d_in[5];
  float* out = (float*)d_out;

  char* ws = (char*)d_ws;
  size_t off = 0;
  auto alloc = [&](size_t bytes) -> char* {
    char* p = ws + off;
    off = (off + bytes + 255) & ~(size_t)255;
    return p;
  };
  const size_t NTOK = (size_t)NB * SEQ;  // 4096
  __bf16* xbf   = (__bf16*)alloc(NTOK * EMB * 2);
  __bf16* wbfT  = (__bf16*)alloc((size_t)EMB * NH * HD * 2);   // [N][K]
  __bf16* wobfT = (__bf16*)alloc((size_t)TOPK * HD * EMB * 2); // [N][K]
  __bf16* tmp   = (__bf16*)alloc(NTOK * NH * HD * 2);
  __bf16* qg    = (__bf16*)alloc(NTOK * TOPK * HD * 2);
  __bf16* kg    = (__bf16*)alloc(NTOK * TOPK * HD * 2);
  __bf16* vg    = (__bf16*)alloc(NTOK * TOPK * HD * 2);
  __bf16* attn  = (__bf16*)alloc(NTOK * TOPK * HD * 2);
  int*    idx   = (int*)alloc(NTOK * TOPK * 4);
  float*  wts   = (float*)alloc(NTOK * TOPK * 4);
  float*  acc   = (float*)alloc((2 * NB * NH + 1) * 4);
  float*  pacc  = acc;
  float*  facc  = acc + NB * NH;
  float*  eacc  = acc + 2 * NB * NH;

  // reset accumulators (idempotent across graph replays)
  zero_f32<<<1, 256, 0, stream>>>(acc, 2 * NB * NH + 1);

  const int NX = (int)(NTOK * EMB);
  f32_to_bf16<<<(NX + 255) / 256, 256, 0, stream>>>(x, xbf, NX);

  // router + aux loss
  router_kernel<<<(int)NTOK, 32, 0, stream>>>(x, Wr, idx, wts, pacc, facc, eacc);
  aux_finalize<<<1, 64, 0, stream>>>(pacc, facc, eacc, out + NTOK * EMB);

  const dim3 gp(NH * HD / 128, (int)NTOK / 128);  // 32 x 32
  const dim3 gt(NH * HD / 32, EMB / 32);          // weight transpose grid
  const int NG = (int)(NTOK * TOPK * HD);

  // Q projection -> gather + RoPE
  transpose_f32_to_bf16<<<gt, 256, 0, stream>>>(Wq, wbfT, EMB, NH * HD);
  gemm_bf16<__bf16><<<gp, 256, 0, stream>>>(xbf, wbfT, tmp, (int)NTOK, NH * HD, EMB);
  gather_heads<true><<<(NG + 255) / 256, 256, 0, stream>>>(tmp, idx, qg);

  // K projection -> gather + RoPE
  transpose_f32_to_bf16<<<gt, 256, 0, stream>>>(Wk, wbfT, EMB, NH * HD);
  gemm_bf16<__bf16><<<gp, 256, 0, stream>>>(xbf, wbfT, tmp, (int)NTOK, NH * HD, EMB);
  gather_heads<true><<<(NG + 255) / 256, 256, 0, stream>>>(tmp, idx, kg);

  // V projection -> gather (transposed)
  transpose_f32_to_bf16<<<gt, 256, 0, stream>>>(Wv, wbfT, EMB, NH * HD);
  gemm_bf16<__bf16><<<gp, 256, 0, stream>>>(xbf, wbfT, tmp, (int)NTOK, NH * HD, EMB);
  gather_v<<<(NG + 255) / 256, 256, 0, stream>>>(tmp, idx, vg);

  // attention
  const dim3 ga(SEQ / 128, TOPK, NB);
  flash_attn<<<ga, 256, 0, stream>>>(qg, kg, vg, wts, attn);

  // output projection (f32 out, directly into d_out)
  transpose_f32_to_bf16<<<dim3(EMB / 32, (TOPK * HD) / 32), 256, 0, stream>>>(
      Wo, wobfT, TOPK * HD, EMB);
  const dim3 go(EMB / 128, (int)NTOK / 128);  // 8 x 32
  gemm_bf16<float><<<go, 256, 0, stream>>>(attn, wobfT, out, (int)NTOK, EMB,
                                           TOPK * HD);
}
